// SphericalVQ_73761768341895
// MI455X (gfx1250) — matrix operand order repaired and checked by reading, not compiled, not run on hardware
//
#include <hip/hip_runtime.h>
#include <hip/hip_bf16.h>

typedef __attribute__((ext_vector_type(16))) _Float16 v16h;
typedef __attribute__((ext_vector_type(8)))  float    v8f;

// Problem constants (from reference setup_inputs)
#define BB        4
#define CCH       64
#define SSP       65536          // nz*nt*nR = 16*64*64
#define NROWS     262144         // BB*SSP
#define KCB       1024           // codebook entries (rows 1..1024; row 0 = padding)

#define ROWS_PER_BLK 128         // 4 waves x 32 rows
#define XS        65             // fp32 LDS row stride (dwords) - bank-conflict free
#define X16S      72             // f16  LDS row stride (halfs)  - 36-dword stride, conflict free, 16B aligned

union FragU { v16h v; uint4 q[2]; };

// ---------------------------------------------------------------------------
// Kernel 1: L2-normalize codebook rows -> fp32 table (for gather) + f16 table
// (for WMMA B fragments). One wave per row; shfl_xor reduction.
// ---------------------------------------------------------------------------
__global__ void vq_wnorm_kernel(const float* __restrict__ w,
                                float* __restrict__ wnf,
                                _Float16* __restrict__ wnh) {
    const int row  = blockIdx.x * 8 + (threadIdx.x >> 5);
    const int lane = threadIdx.x & 31;
    if (row > KCB) return;
    const float* wr = w + (size_t)row * CCH;
    float v0 = wr[lane];
    float v1 = wr[lane + 32];
    float ss = v0 * v0 + v1 * v1;
    #pragma unroll
    for (int m = 16; m >= 1; m >>= 1) ss += __shfl_xor(ss, m, 32);
    const float rn = 1.0f / fmaxf(sqrtf(ss), 1e-12f);
    const float n0 = v0 * rn, n1 = v1 * rn;
    wnf[(size_t)row * CCH + lane]      = n0;
    wnf[(size_t)row * CCH + lane + 32] = n1;
    wnh[(size_t)row * CCH + lane]      = (_Float16)n0;
    wnh[(size_t)row * CCH + lane + 32] = (_Float16)n1;
}

// Load B fragments (kb0 + kb1) for one 16-column codebook tile.
// B 32x16 layout: lane n = code row; lane<16 K {0..7,16..23}, lane>=16 {8..15,24..31}.
#define LOAD_B(tile, F0, F1) do {                                             \
    const _Float16* wp_ = wnh +                                               \
        ((size_t)(((tile) & 63) * 16 + 1 + l15)) * CCH + hsel;                \
    (F0).q[0] = *(const uint4*)(wp_);                                         \
    (F0).q[1] = *(const uint4*)(wp_ + 16);                                    \
    (F1).q[0] = *(const uint4*)(wp_ + 32);                                    \
    (F1).q[1] = *(const uint4*)(wp_ + 48);                                    \
} while (0)

// Two independent accumulator chains (M-tile 0 and 1) sharing B fragments:
// interleaved WMMAs hide the D->C RAW hazard, B reuse halves codebook traffic.
#define STEP(jj, F0, F1) do {                                                 \
    v8f acc0 = {}, acc1 = {};                                                 \
    acc0 = __builtin_amdgcn_wmma_f32_16x16x32_f16(false, a00.v, false,        \
              (F0).v, (short)0, acc0, false, false);                          \
    acc1 = __builtin_amdgcn_wmma_f32_16x16x32_f16(false, a10.v, false,        \
              (F0).v, (short)0, acc1, false, false);                          \
    acc0 = __builtin_amdgcn_wmma_f32_16x16x32_f16(false, a01.v, false,        \
              (F1).v, (short)0, acc0, false, false);                          \
    acc1 = __builtin_amdgcn_wmma_f32_16x16x32_f16(false, a11.v, false,        \
              (F1).v, (short)0, acc1, false, false);                          \
    const int col_ = (jj) * 16 + colbase;                                     \
    _Pragma("unroll")                                                         \
    for (int i = 0; i < 8; ++i) {                                             \
        if (acc0[i] > best0[i]) { best0[i] = acc0[i]; bidx0[i] = col_; }      \
        if (acc1[i] > best1[i]) { best1[i] = acc1[i]; bidx1[i] = col_; }      \
    }                                                                         \
} while (0)

// ---------------------------------------------------------------------------
// Kernel 2: per-128-row tile: normalize x, WMMA score vs all 1024 codes,
// argmax, gather codeword, loss, transposed masked write-out.
// ---------------------------------------------------------------------------
__global__ __launch_bounds__(128)
void vq_main_kernel(const float* __restrict__ x,
                    const float* __restrict__ mask,
                    const float* __restrict__ wnf,
                    const _Float16* __restrict__ wnh,
                    float* __restrict__ qout,
                    float* __restrict__ lossout,
                    float* __restrict__ idxout) {
    __shared__ float    xf[ROWS_PER_BLK * XS];
    __shared__ _Float16 xh[ROWS_PER_BLK * X16S];
    __shared__ int      lidx[ROWS_PER_BLK];

    const int t  = threadIdx.x;           // 0..127, one thread per row
    const int n0 = blockIdx.x * ROWS_PER_BLK;
    const int b  = n0 / SSP;              // tiles never straddle batch (SSP % 128 == 0)
    const int s0 = n0 % SSP;
    const float* xb = x + (size_t)b * CCH * SSP + s0;

    // ---- load x tile [128 rows x 64 ch], coalesced 512B per channel slice ----
    #pragma unroll 4
    for (int c = 0; c < CCH; ++c)
        xf[t * XS + c] = xb[(size_t)c * SSP + t];

    // ---- per-row L2 normalize (own row, no barrier needed yet) ----
    {
        float ss = 0.f;
        float* r = &xf[t * XS];
        #pragma unroll
        for (int c = 0; c < CCH; ++c) { const float v = r[c]; ss += v * v; }
        const float rn = 1.0f / fmaxf(sqrtf(ss), 1e-12f);
        _Float16* rh = &xh[t * X16S];
        #pragma unroll
        for (int c = 0; c < CCH; ++c) {
            const float v = r[c] * rn;
            r[c]  = v;
            rh[c] = (_Float16)v;
        }
    }
    __syncthreads();

    // ---- WMMA score + argmax: 4 waves x 32 rows (two 16-row A tiles) ----
    const int wave   = t >> 5;
    const int lane   = t & 31;
    const int m0     = wave * 32;
    const int l15    = lane & 15;
    const int hsel   = (lane >> 4) * 8;   // A: halfs offset (lanes 16-31 take K hi-half)
    const int hsel16 = (lane >> 4) * 16;  // unused (B loads use hsel)

    // A fragments (16x32 f16 layout): lane<16 K {0..7,16..23}, lane>=16 {8..15,24..31}
    FragU a00, a01, a10, a11;
    {
        const _Float16* rp0 = &xh[(m0 + l15) * X16S];
        const _Float16* rp1 = &xh[(m0 + 16 + l15) * X16S];
        a00.q[0] = *(const uint4*)(rp0 + hsel);
        a00.q[1] = *(const uint4*)(rp0 + 16 + hsel);
        a01.q[0] = *(const uint4*)(rp0 + 32 + hsel);
        a01.q[1] = *(const uint4*)(rp0 + 48 + hsel);
        a10.q[0] = *(const uint4*)(rp1 + hsel);
        a10.q[1] = *(const uint4*)(rp1 + 16 + hsel);
        a11.q[0] = *(const uint4*)(rp1 + 32 + hsel);
        a11.q[1] = *(const uint4*)(rp1 + 48 + hsel);
    }

    float best0[8], best1[8];
    int   bidx0[8], bidx1[8];
    #pragma unroll
    for (int i = 0; i < 8; ++i) {
        best0[i] = -3.4e38f; bidx0[i] = 1;
        best1[i] = -3.4e38f; bidx1[i] = 1;
    }
    const int colbase = l15 + 1;          // codebook row = tile*16 + lane15 + 1

    // Software-pipelined loop: prefetch B for tile j+1 while computing tile j.
    FragU bA0, bA1, bB0, bB1;
    LOAD_B(0, bA0, bA1);
    for (int j = 0; j < 64; j += 2) {
        LOAD_B(j + 1, bB0, bB1);
        STEP(j, bA0, bA1);
        LOAD_B(j + 2, bA0, bA1);          // (j+2)&63 wrap keeps last prefetch in-bounds
        STEP(j + 1, bB0, bB1);
    }

    // reduce across the 16 lanes sharing each row set (xor within 16-lane halves)
    #pragma unroll
    for (int m = 8; m >= 1; m >>= 1) {
        #pragma unroll
        for (int i = 0; i < 8; ++i) {
            float ob; int oi;
            ob = __shfl_xor(best0[i], m, 32);
            oi = __shfl_xor(bidx0[i], m, 32);
            if (ob > best0[i] || (ob == best0[i] && oi < bidx0[i])) {
                best0[i] = ob; bidx0[i] = oi;
            }
            ob = __shfl_xor(best1[i], m, 32);
            oi = __shfl_xor(bidx1[i], m, 32);
            if (ob > best1[i] || (ob == best1[i] && oi < bidx1[i])) {
                best1[i] = ob; bidx1[i] = oi;
            }
        }
    }
    // C/D layout: VGPR i -> M=i (lanes 0-15), M=8+i (lanes 16-31)
    if (l15 == 0) {
        const int mrow = m0 + (lane >> 4) * 8;
        #pragma unroll
        for (int i = 0; i < 8; ++i) {
            lidx[mrow + i]      = bidx0[i];
            lidx[mrow + 16 + i] = bidx1[i];
        }
    }
    __syncthreads();

    // ---- outputs: gather q = wn[idx], loss, transposed masked q_out ----
    {
        const float mv = mask[(size_t)b * SSP + s0 + t];
        const int   id = (mv > 0.f) ? lidx[t] : 0;
        const float* qrow = wnf + (size_t)id * CCH;   // L2-resident gather
        float* outb = qout + (size_t)b * CCH * SSP + s0;
        float ssd = 0.f;
        #pragma unroll 4
        for (int c = 0; c < CCH; ++c) {
            const float qv = qrow[c];
            const float d  = qv - xf[t * XS + c];
            ssd += d * d;
            outb[(size_t)c * SSP + t] = qv * mv;      // q_out = q * mask (straight-through)
        }
        // vq_loss = (1 + 0.25) * mean_c((q - xn)^2), masked
        lossout[n0 + t] = (mv > 0.f) ? (1.25f / (float)CCH) * ssd : 0.f;
        idxout[n0 + t]  = (float)id;
    }
}

extern "C" void kernel_launch(void* const* d_in, const int* in_sizes, int n_in,
                              void* d_out, int out_size, void* d_ws, size_t ws_size,
                              hipStream_t stream) {
    const float* inputs = (const float*)d_in[0];   // [B,C,nz,nt,nR]
    const float* maskl  = (const float*)d_in[1];   // [B,1,nz,nt,nR]
    const float* weight = (const float*)d_in[2];   // [K+1,C]

    // workspace: fp32 normalized table, then f16 table (16B-aligned offsets)
    float*    wnf = (float*)d_ws;                                  // (K+1)*C floats
    _Float16* wnh = (_Float16*)((char*)d_ws + (size_t)(KCB + 1) * CCH * sizeof(float));

    // d_out = [q_out: B*C*S][loss: B*S][idx: B*S] floats
    float* qout    = (float*)d_out;
    float* lossout = qout + (size_t)BB * CCH * SSP;
    float* idxout  = lossout + (size_t)NROWS;

    vq_wnorm_kernel<<<(KCB + 1 + 7) / 8, 256, 0, stream>>>(weight, wnf, wnh);
    vq_main_kernel<<<NROWS / ROWS_PER_BLK, 128, 0, stream>>>(
        inputs, maskl, wnf, wnh, qout, lossout, idxout);
}
